// MultiHeadSelfAttention_19164144075432
// MI455X (gfx1250) — compile-verified, matrix operand order checked by reading, not run on hardware
//
#include <hip/hip_runtime.h>

// ---------------------------------------------------------------------------
// MHA forward for MI455X (gfx1250, wave32, WMMA).
//   q,k,v proj GEMMs (bf16 WMMA, f32 accum) -> flash attention (bf16 WMMA,
//   f32 online softmax) -> two output-projection GEMMs with bias.
// All matrix math uses v_wmma_f32_16x16x32_bf16. Flash formulation keeps the
// 512MB score tensor out of HBM entirely (per-head K/V = 1MB, L2-resident).
// Round 3: batch all LDS fragment loads ahead of the WMMA chains in the flash
// kernel so the compiler can issue partial s_wait_dscnt and overlap the LDS
// pipe with the matrix pipe (as it already does in the GEMM kernel).
// ---------------------------------------------------------------------------

typedef __attribute__((ext_vector_type(16))) __bf16 bf16x16;
typedef __attribute__((ext_vector_type(8)))  float  f32x8;

union FragAB { bf16x16 v; unsigned int u[8]; unsigned short h[16]; };
union FragC  { f32x8   v; float f[8]; };

__device__ __forceinline__ unsigned short f32_to_bf16(float x) {
  union { float f; unsigned int u; } c; c.f = x;
  unsigned int r = c.u + 0x7FFFu + ((c.u >> 16) & 1u);   // round-to-nearest-even
  return (unsigned short)(r >> 16);
}

// ISA 7.12.2 16-bit A-matrix layout: lane holds one row (M = lane&15); its 16
// bf16 elements sit at K offsets {0..7}+{16..23}, plus +8 for lanes>=16.
// K pairs are contiguous -> 8 dword reads (compiler fuses to ds_load_b128).
// p must already include the lane's row base and the (lane>=16 ? 8 : 0)
// offset. B fragments assumed symmetric with N in place of M.
__device__ __forceinline__ void load_frag8(FragAB& fr, const unsigned short* p) {
#pragma unroll
  for (int j = 0; j < 8; ++j) {
    int k = (j < 4) ? (2 * j) : (8 + 2 * j);   // 0,2,4,6,16,18,20,22
    fr.u[j] = *(const unsigned int*)(p + k);
  }
}

// ---------------------------------------------------------------------------
// C[M,N] = A[M,K](f32) * B[K,N](f32) [+ bias], staged to bf16 in LDS,
// computed with v_wmma_f32_16x16x32_bf16. BM=BN=128, BK=32.
// 256 threads = 8 waves in a 4x2 grid; each wave owns a 32x64 C sub-tile
// (2x4 = eight 16x16 WMMA accumulators).
// ---------------------------------------------------------------------------
template <bool OUT_BF16, bool HAS_BIAS>
__global__ __launch_bounds__(256) void gemm_f32in_wmma(
    const float* __restrict__ A, const float* __restrict__ B,
    const float* __restrict__ bias, void* __restrict__ C,
    int M, int N, int K)
{
  __shared__ unsigned short Asl[128 * 32];   // [m][k] bf16
  __shared__ unsigned short Bsl[128 * 32];   // [n][k] bf16 (transposed)

  const int tid  = threadIdx.x;
  const int lane = tid & 31;
  const int wave = tid >> 5;
  const int wm   = wave >> 1;       // 0..3 (32 rows each)
  const int wn   = wave & 1;        // 0..1 (64 cols each)
  const bool hi  = lane >= 16;
  const int lr   = lane & 15;

  const int m0 = blockIdx.y * 128;
  const int n0 = blockIdx.x * 128;

  FragC acc[2][4];
#pragma unroll
  for (int mt = 0; mt < 2; ++mt)
#pragma unroll
    for (int nt = 0; nt < 4; ++nt)
#pragma unroll
      for (int r = 0; r < 8; ++r) acc[mt][nt].f[r] = 0.f;

  for (int k0 = 0; k0 < K; k0 += 32) {
    // stage A (128x32 f32 -> bf16, row-major)
#pragma unroll
    for (int it = 0; it < 4; ++it) {
      int idx = tid + it * 256;
      int row = idx >> 3, c4 = idx & 7;
      const float* src = A + (size_t)(m0 + row) * K + k0 + c4 * 4;
      float4 v = *(const float4*)src;
      if (k0 + 32 < K) __builtin_prefetch(src + 32, 0, 0);   // global_prefetch_b8
      unsigned short* d = Asl + row * 32 + c4 * 4;
      d[0] = f32_to_bf16(v.x); d[1] = f32_to_bf16(v.y);
      d[2] = f32_to_bf16(v.z); d[3] = f32_to_bf16(v.w);
    }
    // stage B (32x128 f32 -> bf16, transposed to [n][k])
#pragma unroll
    for (int it = 0; it < 4; ++it) {
      int idx = tid + it * 256;
      int row = idx >> 5, c4 = idx & 31;
      const float* src = B + (size_t)(k0 + row) * N + n0 + c4 * 4;
      float4 v = *(const float4*)src;
      if (k0 + 32 < K) __builtin_prefetch(src + (size_t)32 * N, 0, 0);
      Bsl[(c4 * 4 + 0) * 32 + row] = f32_to_bf16(v.x);
      Bsl[(c4 * 4 + 1) * 32 + row] = f32_to_bf16(v.y);
      Bsl[(c4 * 4 + 2) * 32 + row] = f32_to_bf16(v.z);
      Bsl[(c4 * 4 + 3) * 32 + row] = f32_to_bf16(v.w);
    }
    __syncthreads();

    FragAB a[2], b[4];
#pragma unroll
    for (int mt = 0; mt < 2; ++mt)
      load_frag8(a[mt], Asl + (wm * 32 + mt * 16 + lr) * 32 + (hi ? 8 : 0));
#pragma unroll
    for (int nt = 0; nt < 4; ++nt)
      load_frag8(b[nt], Bsl + (wn * 64 + nt * 16 + lr) * 32 + (hi ? 8 : 0));
#pragma unroll
    for (int mt = 0; mt < 2; ++mt)
#pragma unroll
      for (int nt = 0; nt < 4; ++nt)
        acc[mt][nt].v = __builtin_amdgcn_wmma_f32_16x16x32_bf16(
            false, a[mt].v, false, b[nt].v, (short)0, acc[mt][nt].v, false, false);
    __syncthreads();
  }

  // store: C/D layout -> row = r + 8*hi, col = lane&15 per 16x16 tile.
  // Fully specialized: no runtime branches in the epilogue.
#pragma unroll
  for (int mt = 0; mt < 2; ++mt)
#pragma unroll
    for (int nt = 0; nt < 4; ++nt) {
      int col = n0 + wn * 64 + nt * 16 + lr;
      float bv = 0.f;
      if (HAS_BIAS) bv = bias[col];
#pragma unroll
      for (int r = 0; r < 8; ++r) {
        int row = m0 + wm * 32 + mt * 16 + r + (hi ? 8 : 0);
        float val = acc[mt][nt].f[r] + bv;
        if (OUT_BF16) ((unsigned short*)C)[(size_t)row * N + col] = f32_to_bf16(val);
        else          ((float*)C)[(size_t)row * N + col] = val;
      }
    }
}

// ---------------------------------------------------------------------------
// Flash attention: one block = (64 queries, 1 head), 128 threads = 4 waves,
// each wave owns 16 query rows. Streams 64-key tiles: S = Q*K^T (WMMA),
// online softmax in f32 (shuffle reductions over 16-lane halves per the C/D
// VGPR layout), P bounced through per-wave LDS for D->A layout conversion
// (in-wave s_wait_dscnt, no barrier needed), O += P*V (WMMA).
// All 8 B-fragments of a tile are loaded before the WMMA chain so DS loads
// and matrix ops overlap (partial DScnt waits).
// ---------------------------------------------------------------------------
__global__ __launch_bounds__(128) void flash_attn_bf16wmma(
    const unsigned short* __restrict__ Qb, const unsigned short* __restrict__ Kb,
    const unsigned short* __restrict__ Vb, float* __restrict__ O)
{
  __shared__ unsigned short Kt[64 * 64];        // [key][dk]
  __shared__ unsigned short Vt[64 * 64];        // [dv][key] (transposed)
  __shared__ unsigned short Pst[4 * 16 * 64];   // per-wave P staging

  const int tid  = threadIdx.x;
  const int lane = tid & 31;
  const int wave = tid >> 5;
  const bool hi  = lane >= 16;
  const int lr   = lane & 15;
  const int h    = blockIdx.y;
  const int q0   = blockIdx.x * 64 + wave * 16;

  // Q A-fragments, dk split 0..31 / 32..63 — resident for the whole kernel
  FragAB qf[2];
#pragma unroll
  for (int ks = 0; ks < 2; ++ks)
    load_frag8(qf[ks], Qb + (size_t)(q0 + lr) * 512 + h * 64 + ks * 32 + (hi ? 8 : 0));

  float mrow[8], lsum[8];
  FragC o[4];
#pragma unroll
  for (int r = 0; r < 8; ++r) { mrow[r] = -3.0e38f; lsum[r] = 0.f; }
#pragma unroll
  for (int nt = 0; nt < 4; ++nt)
#pragma unroll
    for (int r = 0; r < 8; ++r) o[nt].f[r] = 0.f;

  for (int kb = 0; kb < 64; ++kb) {
    // stage K tile (row-major) and V tile (transposed); prefetch next block
#pragma unroll
    for (int it = 0; it < 4; ++it) {
      int idx = tid + it * 128;
      int row = idx >> 3, q4 = idx & 7;
      const unsigned short* ksrc = Kb + (size_t)(kb * 64 + row) * 512 + h * 64 + q4 * 8;
      const unsigned short* vsrc = Vb + (size_t)(kb * 64 + row) * 512 + h * 64 + q4 * 8;
      uint4 kv = *(const uint4*)ksrc;
      *(uint4*)(Kt + row * 64 + q4 * 8) = kv;
      union { uint4 q; unsigned short s[8]; } vv;
      vv.q = *(const uint4*)vsrc;
      if (kb < 63) {                                 // global_prefetch_b8
        __builtin_prefetch(ksrc + (size_t)64 * 512, 0, 0);
        __builtin_prefetch(vsrc + (size_t)64 * 512, 0, 0);
      }
#pragma unroll
      for (int e = 0; e < 8; ++e) Vt[(q4 * 8 + e) * 64 + row] = vv.s[e];
    }
    __syncthreads();

    // ---- S = Q K^T * (1/sqrt(64)): load all 8 K fragments, then 8 WMMAs ----
    FragAB kf[4][2];
#pragma unroll
    for (int nt = 0; nt < 4; ++nt)
#pragma unroll
      for (int ks = 0; ks < 2; ++ks)
        load_frag8(kf[nt][ks], Kt + (nt * 16 + lr) * 64 + ks * 32 + (hi ? 8 : 0));

    FragC s[4];
#pragma unroll
    for (int nt = 0; nt < 4; ++nt)
#pragma unroll
      for (int r = 0; r < 8; ++r) s[nt].f[r] = 0.f;
#pragma unroll
    for (int ks = 0; ks < 2; ++ks)
#pragma unroll
      for (int nt = 0; nt < 4; ++nt)
        s[nt].v = __builtin_amdgcn_wmma_f32_16x16x32_bf16(
            false, qf[ks].v, false, kf[nt][ks].v, (short)0, s[nt].v, false, false);
#pragma unroll
    for (int nt = 0; nt < 4; ++nt)
#pragma unroll
      for (int r = 0; r < 8; ++r) s[nt].f[r] *= 0.125f;

    // online softmax: row stats per C/D-layout row (VGPR r <-> row r / r+8)
#pragma unroll
    for (int r = 0; r < 8; ++r) {
      float mx = s[0].f[r];
#pragma unroll
      for (int nt = 1; nt < 4; ++nt) mx = fmaxf(mx, s[nt].f[r]);
      mx = fmaxf(mx, __shfl_xor(mx, 1));
      mx = fmaxf(mx, __shfl_xor(mx, 2));
      mx = fmaxf(mx, __shfl_xor(mx, 4));
      mx = fmaxf(mx, __shfl_xor(mx, 8));     // masks <16: stays in lane-half
      float mnew  = fmaxf(mrow[r], mx);
      float alpha = __expf(mrow[r] - mnew);
      mrow[r] = mnew;
      float rs = 0.f;
#pragma unroll
      for (int nt = 0; nt < 4; ++nt) {
        float p = __expf(s[nt].f[r] - mnew);
        s[nt].f[r] = p; rs += p;
      }
      rs += __shfl_xor(rs, 1);
      rs += __shfl_xor(rs, 2);
      rs += __shfl_xor(rs, 4);
      rs += __shfl_xor(rs, 8);
      lsum[r] = lsum[r] * alpha + rs;
#pragma unroll
      for (int nt = 0; nt < 4; ++nt) o[nt].f[r] *= alpha;
    }

    // P: D-layout f32 -> bf16 A-layout via per-wave LDS bounce
    unsigned short* P = Pst + wave * 1024;
#pragma unroll
    for (int nt = 0; nt < 4; ++nt)
#pragma unroll
      for (int r = 0; r < 8; ++r)
        P[(r + (hi ? 8 : 0)) * 64 + nt * 16 + lr] = f32_to_bf16(s[nt].f[r]);
    asm volatile("s_wait_dscnt 0" ::: "memory");   // in-wave DS RAW ordering

    // ---- O += P V: load P + all 8 V fragments, then 8 WMMAs ----
    FragAB pf[2];
#pragma unroll
    for (int ks = 0; ks < 2; ++ks)
      load_frag8(pf[ks], P + lr * 64 + ks * 32 + (hi ? 8 : 0));

    FragAB vf[4][2];
#pragma unroll
    for (int nt = 0; nt < 4; ++nt)
#pragma unroll
      for (int ks = 0; ks < 2; ++ks)
        load_frag8(vf[nt][ks], Vt + (nt * 16 + lr) * 64 + ks * 32 + (hi ? 8 : 0));

#pragma unroll
    for (int ks = 0; ks < 2; ++ks)
#pragma unroll
      for (int nt = 0; nt < 4; ++nt)
        o[nt].v = __builtin_amdgcn_wmma_f32_16x16x32_bf16(
            false, pf[ks].v, false, vf[nt][ks].v, (short)0, o[nt].v, false, false);

    __syncthreads();   // protect Kt/Vt before next tile stage
  }

  // normalize and store O[4096][512] f32
#pragma unroll
  for (int r = 0; r < 8; ++r) {
    float inv = 1.f / lsum[r];
    int row = q0 + r + (hi ? 8 : 0);
#pragma unroll
    for (int nt = 0; nt < 4; ++nt)
      O[(size_t)row * 512 + h * 64 + nt * 16 + lr] = o[nt].f[r] * inv;
  }
}

// ---------------------------------------------------------------------------
extern "C" void kernel_launch(void* const* d_in, const int* in_sizes, int n_in,
                              void* d_out, int out_size, void* d_ws, size_t ws_size,
                              hipStream_t stream) {
  (void)in_sizes; (void)n_in; (void)out_size; (void)ws_size;
  const float* queries = (const float*)d_in[0];
  const float* keys    = (const float*)d_in[1];
  const float* values  = (const float*)d_in[2];
  const float* Wq = (const float*)d_in[3];
  const float* Wk = (const float*)d_in[4];
  const float* Wv = (const float*)d_in[5];
  const float* Wo = (const float*)d_in[6];
  const float* bo = (const float*)d_in[7];

  const int M = 4096, N = 512, K = 512;

  // workspace: Qb/Kb/Vb bf16 (3x4MB), O f32 (8MB), Y1 f32 (8MB) = 28MB
  unsigned short* Qb = (unsigned short*)d_ws;
  unsigned short* Kb = Qb + (size_t)M * N;
  unsigned short* Vb = Kb + (size_t)M * N;
  float* O  = (float*)(Vb + (size_t)M * N);
  float* Y1 = O + (size_t)M * N;

  dim3 ggrid(N / 128, M / 128);   // (4, 32)
  gemm_f32in_wmma<true,  false><<<ggrid, 256, 0, stream>>>(queries, Wq, nullptr, Qb, M, N, K);
  gemm_f32in_wmma<true,  false><<<ggrid, 256, 0, stream>>>(keys,    Wk, nullptr, Kb, M, N, K);
  gemm_f32in_wmma<true,  false><<<ggrid, 256, 0, stream>>>(values,  Wv, nullptr, Vb, M, N, K);
  flash_attn_bf16wmma<<<dim3(M / 64, 8), 128, 0, stream>>>(Qb, Kb, Vb, O);
  gemm_f32in_wmma<false, true ><<<ggrid, 256, 0, stream>>>(O,  Wo, bo, Y1, M, N, K);
  gemm_f32in_wmma<false, true ><<<ggrid, 256, 0, stream>>>(Y1, Wo, bo, (float*)d_out, M, N, K);
}
